// TripletLoss_WRT_20263655702978
// MI455X (gfx1250) — matrix operand order repaired and checked by reading, not compiled, not run on hardware
//
#include <hip/hip_runtime.h>

#define MARGIN 0.3f
#define N_ROWS 8192
#define DIM    256

typedef __bf16        v16bf __attribute__((ext_vector_type(16)));
typedef float         v8f   __attribute__((ext_vector_type(8)));
typedef float         f32x4 __attribute__((ext_vector_type(4)));
typedef unsigned int  u32x4 __attribute__((ext_vector_type(4)));

// ---------- bf16 split helpers (bit math, no __bf16 arithmetic) ----------
__device__ __forceinline__ unsigned short f32_to_bf16_rne(float f) {
    unsigned int u = __float_as_uint(f);
    unsigned int r = u + 0x7FFFu + ((u >> 16) & 1u);
    return (unsigned short)(r >> 16);
}
__device__ __forceinline__ float bf16_to_f32(unsigned short h) {
    return __uint_as_float(((unsigned int)h) << 16);
}

// Load a 16-element bf16 fragment slice as two 16B vectors, bit-cast to v16bf.
__device__ __forceinline__ v16bf load_frag(const unsigned short* p) {
    u32x4 a = *(const u32x4*)(p);
    u32x4 b = *(const u32x4*)(p + 8);
    v16bf r;
    __builtin_memcpy(&r, &a, 16);
    __builtin_memcpy((char*)&r + 16, &b, 16);
    return r;
}

// ---------------- Kernel 1: L2-normalize, emit bf16 hi/lo split ----------------
// 1 wave32 per row; block = 256 threads = 8 rows.
__global__ void normalize_kernel(const float* __restrict__ X,
                                 unsigned short* __restrict__ Xhi,
                                 unsigned short* __restrict__ Xlo,
                                 float* __restrict__ sq) {
    const int row  = blockIdx.x * 8 + (threadIdx.x >> 5);
    const int lane = threadIdx.x & 31;
    const float* xr = X + (size_t)row * DIM;

    f32x4 a = *(const f32x4*)(xr + lane * 8);
    f32x4 b = *(const f32x4*)(xr + lane * 8 + 4);
    float s = a.x*a.x + a.y*a.y + a.z*a.z + a.w*a.w
            + b.x*b.x + b.y*b.y + b.z*b.z + b.w*b.w;
    #pragma unroll
    for (int m = 16; m >= 1; m >>= 1) s += __shfl_xor(s, m, 32);

    const float nrm = sqrtf(s) + 1e-12f;
    const float inv = 1.0f / nrm;
    if (lane == 0) sq[row] = s * inv * inv;   // sum(x_norm^2), as reference computes it

    unsigned short hi[8], lo[8];
    float v[8] = {a.x, a.y, a.z, a.w, b.x, b.y, b.z, b.w};
    #pragma unroll
    for (int e = 0; e < 8; ++e) {
        float f = v[e] * inv;
        hi[e] = f32_to_bf16_rne(f);
        lo[e] = f32_to_bf16_rne(f - bf16_to_f32(hi[e]));
    }
    u32x4 hpack, lpack;
    __builtin_memcpy(&hpack, hi, 16);
    __builtin_memcpy(&lpack, lo, 16);
    *(u32x4*)(Xhi + (size_t)row * DIM + lane * 8) = hpack;
    *(u32x4*)(Xlo + (size_t)row * DIM + lane * 8) = lpack;
}

// ---------------- Kernel 2: fused Gram tile (split-bf16 WMMA) + hard mining ----
// Block = 128 threads (4 waves) owns a 16-row M-tile.
// Each wave sweeps 128 of the 512 J-tiles, two tiles per iteration (dual acc),
// with the full A strip (hi+lo, 16x256) resident in VGPRs and the B fragments
// double-buffered so chunk c+1's loads overlap chunk c's WMMAs (no WAR stall).
__global__ void __launch_bounds__(128, 1)
gram_mine_kernel(const unsigned short* __restrict__ Xhi,
                 const unsigned short* __restrict__ Xlo,
                 const float* __restrict__ sq,
                 const int* __restrict__ tgt,
                 float* __restrict__ rowLoss) {
    const int iBase = blockIdx.x * 16;
    const int tid   = threadIdx.x;
    const int wave  = tid >> 5;
    const int lane  = tid & 31;
    const int l15   = lane & 15;
    const int half  = lane >> 4;          // lane group selects K sub-range / row half

    // --- Preload A fragments for this M-tile (hi & lo), 8 K-chunks of 32 ---
    // 16-bit A layout: lane l -> row (l&15); element e -> k = 32c + 16*half + e.
    const unsigned short* aHiRow = Xhi + (size_t)(iBase + l15) * DIM;
    const unsigned short* aLoRow = Xlo + (size_t)(iBase + l15) * DIM;
    v16bf Ahi[8], Alo[8];
    #pragma unroll
    for (int c = 0; c < 8; ++c) {
        const int k0 = 32 * c + 16 * half;
        Ahi[c] = load_frag(aHiRow + k0);
        Alo[c] = load_frag(aLoRow + k0);
    }

    // Per-row (m = r + 8*half) metadata, hoisted out of the J loop.
    float sqi[8]; int ti[8];
    #pragma unroll
    for (int r = 0; r < 8; ++r) {
        const int i = iBase + r + 8 * half;
        sqi[r] = sq[i];
        ti[r]  = tgt[i];
    }

    float maxPos[8], minNeg[8];
    #pragma unroll
    for (int r = 0; r < 8; ++r) { maxPos[r] = -__builtin_inff(); minNeg[r] = __builtin_inff(); }

    // Epilogue: distance + hard-positive/negative mining, fused in registers.
    auto mine = [&](const v8f& acc, int j) {
        const float sqj = sq[j];
        const int   tj  = tgt[j];
        #pragma unroll
        for (int r = 0; r < 8; ++r) {
            const int   i  = iBase + r + 8 * half;   // C layout: VGPR r -> row r+8*half
            const float d2 = sqi[r] + sqj - 2.0f * acc[r];
            const float d  = sqrtf(fmaxf(d2, 1e-12f));
            if (ti[r] == tj) {
                if (j != i) maxPos[r] = fmaxf(maxPos[r], d);  // positives exclude diagonal
            } else {
                minNeg[r] = fminf(minNeg[r], d);              // negatives
            }
        }
    };

    // --- Sweep column tiles in pairs; wave-uniform loop (EXEC all-1s at WMMA) ---
    for (int jt0 = wave * 2; jt0 < N_ROWS / 16; jt0 += 8) {
        const int j0 = jt0 * 16 + l15;        // lane's column in tile 0 (B: lane -> N)
        const int j1 = j0 + 16;               // lane's column in tile 1
        const unsigned short* b0h = Xhi + (size_t)j0 * DIM;
        const unsigned short* b0l = Xlo + (size_t)j0 * DIM;
        const unsigned short* b1h = Xhi + (size_t)j1 * DIM;
        const unsigned short* b1l = Xlo + (size_t)j1 * DIM;

        // Prefetch next iteration's B rows toward the WGP (data is L2-resident).
        if (jt0 + 8 < N_ROWS / 16) {
            __builtin_prefetch(b0h + 128 * DIM, 0, 1);
            __builtin_prefetch(b0l + 128 * DIM, 0, 1);
            __builtin_prefetch(b1h + 128 * DIM, 0, 1);
            __builtin_prefetch(b1l + 128 * DIM, 0, 1);
        }

        // Double-buffered B fragments: [buf][tile0.hi, tile0.lo, tile1.hi, tile1.lo]
        v16bf Bb[2][4];
        auto loadB = [&](int c, int buf) {
            const int k0 = 32 * c + 16 * half;
            Bb[buf][0] = load_frag(b0h + k0);
            Bb[buf][1] = load_frag(b0l + k0);
            Bb[buf][2] = load_frag(b1h + k0);
            Bb[buf][3] = load_frag(b1l + k0);
        };

        v8f acc0 = {0.f, 0.f, 0.f, 0.f, 0.f, 0.f, 0.f, 0.f};
        v8f acc1 = {0.f, 0.f, 0.f, 0.f, 0.f, 0.f, 0.f, 0.f};
        loadB(0, 0);
        #pragma unroll
        for (int c = 0; c < 8; ++c) {
            const int cur = c & 1;
            if (c < 7) loadB(c + 1, cur ^ 1);   // issue next chunk while computing
            // split-f32: x.y ~= hi.hi + hi.lo + lo.hi (lo.lo negligible)
            acc0 = __builtin_amdgcn_wmma_f32_16x16x32_bf16(false, Ahi[c], false, Bb[cur][0],
                                                           (short)0, acc0, false, false);
            acc1 = __builtin_amdgcn_wmma_f32_16x16x32_bf16(false, Ahi[c], false, Bb[cur][2],
                                                           (short)0, acc1, false, false);
            acc0 = __builtin_amdgcn_wmma_f32_16x16x32_bf16(false, Ahi[c], false, Bb[cur][1],
                                                           (short)0, acc0, false, false);
            acc1 = __builtin_amdgcn_wmma_f32_16x16x32_bf16(false, Ahi[c], false, Bb[cur][3],
                                                           (short)0, acc1, false, false);
            acc0 = __builtin_amdgcn_wmma_f32_16x16x32_bf16(false, Alo[c], false, Bb[cur][0],
                                                           (short)0, acc0, false, false);
            acc1 = __builtin_amdgcn_wmma_f32_16x16x32_bf16(false, Alo[c], false, Bb[cur][2],
                                                           (short)0, acc1, false, false);
        }

        mine(acc0, j0);
        mine(acc1, j1);
    }

    // --- Reduce across the 16 N-lanes of each half-wave ---
    #pragma unroll
    for (int m = 8; m >= 1; m >>= 1) {
        #pragma unroll
        for (int r = 0; r < 8; ++r) {
            maxPos[r] = fmaxf(maxPos[r], __shfl_xor(maxPos[r], m, 32));
            minNeg[r] = fminf(minNeg[r], __shfl_xor(minNeg[r], m, 32));
        }
    }

    // --- Combine the 4 waves (each covered a disjoint J subset) via LDS ---
    __shared__ float smax[4][16];
    __shared__ float smin[4][16];
    if (l15 == 0) {
        #pragma unroll
        for (int r = 0; r < 8; ++r) {
            smax[wave][r + 8 * half] = maxPos[r];
            smin[wave][r + 8 * half] = minNeg[r];
        }
    }
    __syncthreads();

    if (tid < 16) {
        float ap = fmaxf(fmaxf(smax[0][tid], smax[1][tid]),
                         fmaxf(smax[2][tid], smax[3][tid]));
        float an = fminf(fminf(smin[0][tid], smin[1][tid]),
                         fminf(smin[2][tid], smin[3][tid]));
        if (ap == -__builtin_inff()) ap = 0.0f;          // no positives -> 0
        if (an ==  __builtin_inff()) an = MARGIN + 1.0f; // no negatives -> margin+1
        rowLoss[iBase + tid] = fmaxf(0.0f, ap - an + MARGIN);
    }
}

// ---------------- Kernel 3: deterministic mean reduction ----------------
__global__ void reduce_kernel(const float* __restrict__ rowLoss, float* __restrict__ out) {
    __shared__ float s[256];
    const int t = threadIdx.x;
    float acc = 0.0f;
    for (int k = t; k < N_ROWS; k += 256) acc += rowLoss[k];
    s[t] = acc;
    __syncthreads();
    for (int w = 128; w >= 1; w >>= 1) {
        if (t < w) s[t] += s[t + w];
        __syncthreads();
    }
    if (t == 0) out[0] = s[0] * (1.0f / (float)N_ROWS);
}

extern "C" void kernel_launch(void* const* d_in, const int* in_sizes, int n_in,
                              void* d_out, int out_size, void* d_ws, size_t ws_size,
                              hipStream_t stream) {
    (void)in_sizes; (void)n_in; (void)out_size; (void)ws_size;
    const float* X   = (const float*)d_in[0];
    const int*   tgt = (const int*)d_in[1];
    float*       out = (float*)d_out;

    char* ws = (char*)d_ws;
    unsigned short* Xhi = (unsigned short*)(ws);                                   // 4 MB
    unsigned short* Xlo = (unsigned short*)(ws + (size_t)N_ROWS * DIM * 2);        // 4 MB
    float* sq      = (float*)(ws + (size_t)N_ROWS * DIM * 4);                      // 32 KB
    float* rowLoss = (float*)(ws + (size_t)N_ROWS * DIM * 4 + N_ROWS * 4);         // 32 KB

    normalize_kernel<<<N_ROWS / 8, 256, 0, stream>>>(X, Xhi, Xlo, sq);
    gram_mine_kernel<<<N_ROWS / 16, 128, 0, stream>>>(Xhi, Xlo, sq, tgt, rowLoss);
    reduce_kernel<<<1, 256, 0, stream>>>(rowLoss, out);
}
